// MambaGNNBlock_1133871366250
// MI455X (gfx1250) — matrix operand order repaired and checked by reading, not compiled, not run on hardware
//
#include <hip/hip_runtime.h>
#include <hip/hip_bf16.h>

#define NN      50000
#define NE      800000
#define FDIM    128
#define NCHUNK  500
#define CHUNK   100   // NCHUNK*CHUNK == NN

#define USE_TDM 1     // Tensor Data Mover for A-panel staging (6-arg clang-23 builtin)

typedef __attribute__((ext_vector_type(16))) __bf16       v16bf;
typedef __attribute__((ext_vector_type(8)))  float        v8f;
typedef __attribute__((ext_vector_type(4)))  unsigned int v4u;
typedef __attribute__((ext_vector_type(8)))  int          v8i;
typedef __attribute__((ext_vector_type(4)))  int          v4i;

__device__ __forceinline__ float softplusf(float x) {
    return x > 20.0f ? x : log1pf(__expf(x));
}

__device__ __forceinline__ void atomic_add_f32(float* p, float v) {
    __hip_atomic_fetch_add(p, v, __ATOMIC_RELAXED, __HIP_MEMORY_SCOPE_AGENT);
}

// ---------------------------------------------------------------------------
// Pack W[128,kout] (f32, row-major) into WMMA B-fragment order as bf16:
//   out[(((kk*ntiles + tile)*32 + lane)*16 + e]  with e=2r+j:
//   k = kk*32 + (r<4 ? 2r : 16+2(r-4)) + 8*(lane>>4) + j ; n = tile*16 + (lane&15)
// Each lane of the GEMM then loads its whole fragment as one 32B chunk.
// ---------------------------------------------------------------------------
__global__ __launch_bounds__(256)
void pack_w(const float* __restrict__ W, __bf16* __restrict__ out, int kout)
{
    int idx = blockIdx.x * 256 + threadIdx.x;
    if (idx >= 128 * kout) return;
    int e = idx & 15, t = idx >> 4;
    int lane = t & 31; t >>= 5;
    int ntiles = kout >> 4;
    int tile = t % ntiles, kk = t / ntiles;
    int half = lane >> 4, m = lane & 15;
    int r = e >> 1, j = e & 1;
    int k = kk * 32 + ((r < 4) ? (2 * r) : (16 + 2 * (r - 4))) + 8 * half + j;
    int n = tile * 16 + m;
    out[idx] = (__bf16)W[(size_t)k * kout + n];
}

// ---------------------------------------------------------------------------
// GEMM: C[NN, kout] (+)= A[NN,128] @ W[128,kout]  (bf16 WMMA, f32 accum)
// W passed pre-packed (see pack_w). flags: bit0 = accumulate, bit1 = relu
// grid = (NN/16, kout/128), block = 128 (4 waves; each wave 2 col tiles)
// ---------------------------------------------------------------------------
__global__ __launch_bounds__(128)
void gemm_wmma(const float* __restrict__ A, const __bf16* __restrict__ Wp16,
               const float* __restrict__ bias, float* __restrict__ C,
               int kout, int flags)
{
    const int rowBase = blockIdx.x * 16;
    const int tid  = threadIdx.x;
    const int wave = tid >> 5;
    const int lane = tid & 31;
    const int half = lane >> 4;
    const int m    = lane & 15;
    const int ntiles = kout >> 4;

#if USE_TDM
    // TDM loads the 16x128 f32 A panel with 1-DWORD padding per 128-DWORD row
    // -> LDS pitch = 129 floats (conflict-free strided reads by row m).
    __shared__ float Asf[16 * 129];
    if (wave == 0) {
        unsigned long long ga = (unsigned long long)(uintptr_t)(A + (size_t)rowBase * FDIM);
        unsigned int lds0 = (unsigned int)(uintptr_t)(&Asf[0]);
        v4u g0 = { 1u,                                   // count=1, user desc
                   lds0,                                 // lds_addr  (bits 63:32)
                   (unsigned int)ga,                     // global_addr[31:0]
                   (unsigned int)(ga >> 32) | (2u << 30) // global_addr[56:32] | type=2
                 };
        v8i g1 = { (int)((2u << 16) | (1u << 20) | (6u << 22)), // data_size=4B, pad_en, intv=128dw, amt=1dw
                   (int)(128u << 16),                    // tensor_dim0 = 128  (bits 63:48)
                   (int)(16u  << 16),                    // tensor_dim1 = 16   (bits 95:80)
                   (int)(128u << 16),                    // tile_dim0   = 128  (bits 127:112)
                   16,                                   // tile_dim1   = 16   (bits 143:128)
                   128,                                  // tensor_dim0_stride = 128
                   0, 0 };
        v4i gz4 = { 0, 0, 0, 0 };
        v8i gz8 = { 0, 0, 0, 0, 0, 0, 0, 0 };
        __builtin_amdgcn_tensor_load_to_lds(g0, g1, gz4, gz4, gz8, 0);
        __builtin_amdgcn_s_wait_tensorcnt(0);
    }
    __syncthreads();
#else
    __shared__ __bf16 As[16 * 130];
    for (int idx = tid; idx < 16 * 128; idx += 128) {
        int r = idx >> 7, c = idx & 127;
        As[r * 130 + c] = (__bf16)A[(size_t)(rowBase + r) * FDIM + c];
    }
    __syncthreads();
#endif

    const v16bf* __restrict__ WT = (const v16bf*)Wp16;

    v8f acc0 = {0.f,0.f,0.f,0.f,0.f,0.f,0.f,0.f};
    v8f acc1 = {0.f,0.f,0.f,0.f,0.f,0.f,0.f,0.f};

    #pragma unroll
    for (int kk = 0; kk < 4; ++kk) {          // K = 128 in steps of 32
        const int kb = kk * 32;

        // A fragment (16-bit A 16x32 layout): VGPR r<4 -> K=2r, r>=4 -> 16+2(r-4); hi half +8
        v16bf a;
        #pragma unroll
        for (int r = 0; r < 8; ++r) {
            int k = kb + ((r < 4) ? (2 * r) : (16 + 2 * (r - 4))) + 8 * half;
#if USE_TDM
            a[2 * r]     = (__bf16)Asf[m * 129 + k];
            a[2 * r + 1] = (__bf16)Asf[m * 129 + k + 1];
#else
            a[2 * r]     = As[m * 130 + k];
            a[2 * r + 1] = As[m * 130 + k + 1];
#endif
        }

        #pragma unroll
        for (int t = 0; t < 2; ++t) {
            int tileg = blockIdx.y * 8 + wave * 2 + t;
            v16bf b = WT[(size_t)(kk * ntiles + tileg) * 32 + lane];  // 32B/lane, coalesced
            if (t == 0)
                acc0 = __builtin_amdgcn_wmma_f32_16x16x32_bf16(false, a, false, b,
                                                               (short)0, acc0, false, false);
            else
                acc1 = __builtin_amdgcn_wmma_f32_16x16x32_bf16(false, a, false, b,
                                                               (short)0, acc1, false, false);
        }
    }

    #pragma unroll
    for (int t = 0; t < 2; ++t) {
        int col = (blockIdx.y * 8 + wave * 2 + t) * 16 + m;
        float bv = bias ? bias[col] : 0.0f;
        #pragma unroll
        for (int i = 0; i < 8; ++i) {
            int row = rowBase + i + 8 * half;
            float v = (t == 0 ? acc0[i] : acc1[i]) + bv;
            size_t o = (size_t)row * kout + col;
            if (flags & 1) v += C[o];
            if (flags & 2) v = fmaxf(v, 0.0f);
            C[o] = v;
        }
    }
}

// ---------------------------------------------------------------------------
// Edge kernels
// ---------------------------------------------------------------------------
__global__ __launch_bounds__(256)
void edge_count(const int* __restrict__ ei, float* __restrict__ cnt)
{
    int e = blockIdx.x * 256 + threadIdx.x;
    if (e < NE) atomic_add_f32(&cnt[ei[NE + e]], 1.0f);
}

__global__ __launch_bounds__(256)
void dinv_kernel(const float* __restrict__ cnt, float* __restrict__ dinv)
{
    int i = blockIdx.x * 256 + threadIdx.x;
    if (i < NN) dinv[i] = rsqrtf(cnt[i] + 1.0f);   // +1 self loop
}

// one wave per edge, 4 floats per lane; use_coef: 1 = dinv[src]*dinv[dst], 0 = 1.0
__global__ __launch_bounds__(256)
void edge_agg(const float* __restrict__ feat, const int* __restrict__ ei,
              const float* __restrict__ dinv, float* __restrict__ agg, int use_coef)
{
    int gw   = (blockIdx.x * 256 + threadIdx.x) >> 5;
    int lane = threadIdx.x & 31;
    if (gw >= NE) return;
    int src = ei[gw], dst = ei[NE + gw];
    float coef = use_coef ? dinv[src] * dinv[dst] : 1.0f;
    float4 v = ((const float4*)(feat + (size_t)src * FDIM))[lane];
    float* o = agg + (size_t)dst * FDIM + lane * 4;
    atomic_add_f32(o + 0, v.x * coef);
    atomic_add_f32(o + 1, v.y * coef);
    atomic_add_f32(o + 2, v.z * coef);
    atomic_add_f32(o + 3, v.w * coef);
}

__global__ __launch_bounds__(256)
void gcn_finalize(const float* __restrict__ agg, const float* __restrict__ xw,
                  const float* __restrict__ dinv, const float* __restrict__ b,
                  float* __restrict__ out, int relu)
{
    int idx = blockIdx.x * 256 + threadIdx.x;
    if (idx >= NN * FDIM) return;
    int row = idx >> 7, c = idx & 127;
    float di = dinv[row];
    float v = agg[idx] + xw[idx] * di * di + b[c];
    if (relu) v = fmaxf(v, 0.0f);
    out[idx] = v;
}

__global__ __launch_bounds__(256)
void sage_mean(const float* __restrict__ agg, const float* __restrict__ cnt,
               float* __restrict__ out)
{
    int idx = blockIdx.x * 256 + threadIdx.x;
    if (idx >= NN * FDIM) return;
    out[idx] = agg[idx] / fmaxf(cnt[idx >> 7], 1.0f);
}

// ---------------------------------------------------------------------------
// Chunked parallel scan for s_i = a_i * s_{i-1} + b_i   (per channel c)
// a_i = softplus(delta_i)*A[c];  b_i = softplus(delta_i)*Bm[c]*resid_i
// proj row layout: [delta(128) | Bunused(128) | C(128) | resid(128)]
// ---------------------------------------------------------------------------
__global__ __launch_bounds__(128)
void scan_pass1(const float* __restrict__ proj, const float* __restrict__ Ap,
                const float* __restrict__ Bm, float* __restrict__ cA, float* __restrict__ cB)
{
    int c = threadIdx.x, ch = blockIdx.x;
    int i0 = ch * CHUNK, i1 = i0 + CHUNK;
    float Ac = Ap[c], Bc = Bm[c];
    float aacc = 1.0f, bacc = 0.0f;
    for (int i = i0; i < i1; ++i) {
        const float* p = proj + (size_t)i * 512;
        float sp = softplusf(p[c]);
        float a  = sp * Ac;
        float b  = sp * Bc * p[384 + c];
        bacc = a * bacc + b;
        aacc = a * aacc;
    }
    cA[ch * FDIM + c] = aacc;
    cB[ch * FDIM + c] = bacc;
}

__global__ __launch_bounds__(128)
void scan_pass2(const float* __restrict__ cA, const float* __restrict__ cB,
                float* __restrict__ cS)
{
    int c = threadIdx.x;
    float s = 0.0f;
    for (int ch = 0; ch < NCHUNK; ++ch) {
        cS[ch * FDIM + c] = s;                       // exclusive chunk-entry state
        s = cA[ch * FDIM + c] * s + cB[ch * FDIM + c];
    }
}

__global__ __launch_bounds__(128)
void scan_pass3(const float* __restrict__ proj, const float* __restrict__ Ap,
                const float* __restrict__ Bm, const float* __restrict__ Dp,
                const float* __restrict__ cS, float* __restrict__ out)
{
    int c = threadIdx.x, ch = blockIdx.x;
    int i0 = ch * CHUNK, i1 = i0 + CHUNK;
    float Ac = Ap[c], Bc = Bm[c], Dc = Dp[c];
    float s = cS[ch * FDIM + c];
    for (int i = i0; i < i1; ++i) {
        const float* p = proj + (size_t)i * 512;
        float sp = softplusf(p[c]);
        float resid = p[384 + c];
        s = sp * Ac * s + sp * Bc * resid;
        out[(size_t)i * FDIM + c] = s * p[256 + c] + resid * Dc;
    }
}

// ---------------------------------------------------------------------------
// LayerNorm over last dim (128), one wave per row; optional residual add
// ---------------------------------------------------------------------------
__global__ __launch_bounds__(256)
void layernorm_kernel(const float* __restrict__ in, const float* __restrict__ addsrc,
                      float* __restrict__ out)
{
    int gw   = (blockIdx.x * 256 + threadIdx.x) >> 5;
    int lane = threadIdx.x & 31;
    if (gw >= NN) return;
    float4 v = ((const float4*)(in + (size_t)gw * FDIM))[lane];
    float s = v.x + v.y + v.z + v.w;
    #pragma unroll
    for (int o = 16; o > 0; o >>= 1) s += __shfl_xor(s, o, 32);
    float mu = s * (1.0f / 128.0f);
    float dx = v.x - mu, dy = v.y - mu, dz = v.z - mu, dw = v.w - mu;
    float q = dx * dx + dy * dy + dz * dz + dw * dw;
    #pragma unroll
    for (int o = 16; o > 0; o >>= 1) q += __shfl_xor(q, o, 32);
    float inv = rsqrtf(q * (1.0f / 128.0f) + 1e-5f);
    float4 r = make_float4(dx * inv, dy * inv, dz * inv, dw * inv);
    if (addsrc) {
        float4 a = ((const float4*)(addsrc + (size_t)gw * FDIM))[lane];
        r.x += a.x; r.y += a.y; r.z += a.z; r.w += a.w;
    }
    ((float4*)(out + (size_t)gw * FDIM))[lane] = r;
}

// ---------------------------------------------------------------------------
extern "C" void kernel_launch(void* const* d_in, const int* in_sizes, int n_in,
                              void* d_out, int out_size, void* d_ws, size_t ws_size,
                              hipStream_t stream)
{
    const float* x        = (const float*)d_in[0];
    const int*   ei       = (const int*)  d_in[1];   // [2, NE]: row0 src, row1 dst
    const float* W_gcn1   = (const float*)d_in[2];
    const float* b_gcn1   = (const float*)d_in[3];
    const float* Wp       = (const float*)d_in[4];   // [3,128,512]
    const float* Ap       = (const float*)d_in[5];   // [3,128]
    const float* Bm       = (const float*)d_in[6];
    const float* Dp       = (const float*)d_in[7];
    const float* W_sage_l = (const float*)d_in[8];
    const float* b_sage_l = (const float*)d_in[9];
    const float* W_sage_r = (const float*)d_in[10];
    const float* W_res1   = (const float*)d_in[11];
    const float* b_res1   = (const float*)d_in[12];
    const float* W_res2   = (const float*)d_in[13];
    const float* b_res2   = (const float*)d_in[14];
    const float* W_fuse   = (const float*)d_in[15];  // [256,128]
    const float* b_fuse   = (const float*)d_in[16];
    float* out = (float*)d_out;

    const size_t NF = (size_t)NN * FDIM;
    float* w    = (float*)d_ws;
    float* dinv = w;
    float* cnt  = dinv + NN;
    float* x2   = cnt + NN;
    float* mbuf = x2 + NF;
    float* xw   = mbuf + NF;
    float* agg  = xw + NF;
    float* cA   = agg + NF;
    float* cB   = cA + (size_t)NCHUNK * FDIM;
    float* cS   = cB + (size_t)NCHUNK * FDIM;
    float* proj = cS + (size_t)NCHUNK * FDIM;        // [NN,512], recycled below
    float* xmp  = proj;                              // x_mamba_pre = LN(m3)+x2
    float* xm   = proj + NF;                         // x_mamba (SAGE out)
    float* h1   = proj + 2 * NF;
    float* xgnn = proj + 3 * NF;

    // packed bf16 weights (WMMA fragment order), after the proj region
    __bf16* pw        = (__bf16*)(proj + (size_t)NN * 512);
    __bf16* p_gcn1    = pw;                  // 128*128
    __bf16* p_wp0     = p_gcn1 + 16384;      // 128*512 each
    __bf16* p_wp1     = p_wp0 + 65536;
    __bf16* p_wp2     = p_wp1 + 65536;
    __bf16* p_sage_l  = p_wp2 + 65536;
    __bf16* p_sage_r  = p_sage_l + 16384;
    __bf16* p_res1    = p_sage_r + 16384;
    __bf16* p_res2    = p_res1 + 16384;
    __bf16* p_fuse_t  = p_res2 + 16384;
    __bf16* p_fuse_b  = p_fuse_t + 16384;

    const int EW_BLK = NE / 8;          // wave-per-edge grid (8 waves/block)
    const int EL_BLK = (NE + 255) / 256;
    const int NL_BLK = (NN + 255) / 256;
    const int NF_BLK = (int)((NF + 255) / 256);
    const int LN_BLK = NN / 8;          // 8 rows per 256-thread block
    const dim3 G128(NN / 16, 1), G512(NN / 16, 4);
    const int PK128 = 16384 / 256, PK512 = 65536 / 256;

    // pack all weights to bf16 fragment order (tiny, once per launch)
    pack_w<<<PK128, 256, 0, stream>>>(W_gcn1, p_gcn1, 128);
    pack_w<<<PK512, 256, 0, stream>>>(Wp,                 p_wp0, 512);
    pack_w<<<PK512, 256, 0, stream>>>(Wp + 1 * 128 * 512, p_wp1, 512);
    pack_w<<<PK512, 256, 0, stream>>>(Wp + 2 * 128 * 512, p_wp2, 512);
    pack_w<<<PK128, 256, 0, stream>>>(W_sage_l, p_sage_l, 128);
    pack_w<<<PK128, 256, 0, stream>>>(W_sage_r, p_sage_r, 128);
    pack_w<<<PK128, 256, 0, stream>>>(W_res1, p_res1, 128);
    pack_w<<<PK128, 256, 0, stream>>>(W_res2, p_res2, 128);
    pack_w<<<PK128, 256, 0, stream>>>(W_fuse,             p_fuse_t, 128);
    pack_w<<<PK128, 256, 0, stream>>>(W_fuse + 128 * 128, p_fuse_b, 128);

    // degrees / normalization
    (void)hipMemsetAsync(cnt, 0, NN * sizeof(float), stream);
    edge_count<<<EL_BLK, 256, 0, stream>>>(ei, cnt);
    dinv_kernel<<<NL_BLK, 256, 0, stream>>>(cnt, dinv);

    // x2 = gcn_conv(x, W_gcn1, b_gcn1)
    gemm_wmma<<<G128, 128, 0, stream>>>(x, p_gcn1, nullptr, xw, 128, 0);
    (void)hipMemsetAsync(agg, 0, NF * sizeof(float), stream);
    edge_agg<<<EW_BLK, 256, 0, stream>>>(xw, ei, dinv, agg, 1);
    gcn_finalize<<<NF_BLK, 256, 0, stream>>>(agg, xw, dinv, b_gcn1, x2, 0);

    // 3 Mamba layers: proj GEMM -> chunked scan -> layernorm
    const __bf16* pwp[3] = { p_wp0, p_wp1, p_wp2 };
    const float* mm_in = x2;
    for (int l = 0; l < 3; ++l) {
        gemm_wmma<<<G512, 128, 0, stream>>>(mm_in, pwp[l], nullptr, proj, 512, 0);
        scan_pass1<<<NCHUNK, 128, 0, stream>>>(proj, Ap + l * FDIM, Bm + l * FDIM, cA, cB);
        scan_pass2<<<1, 128, 0, stream>>>(cA, cB, cS);
        scan_pass3<<<NCHUNK, 128, 0, stream>>>(proj, Ap + l * FDIM, Bm + l * FDIM,
                                               Dp + l * FDIM, cS, mbuf);
        if (l < 2) {
            layernorm_kernel<<<LN_BLK, 256, 0, stream>>>(mbuf, nullptr, mbuf);
            mm_in = mbuf;
        } else {
            layernorm_kernel<<<LN_BLK, 256, 0, stream>>>(mbuf, x2, xmp);  // + x2 residual
        }
    }

    // SAGE: mean-agg -> W_l + bias, += xmp @ W_r
    (void)hipMemsetAsync(agg, 0, NF * sizeof(float), stream);
    edge_agg<<<EW_BLK, 256, 0, stream>>>(xmp, ei, dinv, agg, 0);
    sage_mean<<<NF_BLK, 256, 0, stream>>>(agg, cnt, xw);
    gemm_wmma<<<G128, 128, 0, stream>>>(xw,  p_sage_l, b_sage_l, xm, 128, 0);
    gemm_wmma<<<G128, 128, 0, stream>>>(xmp, p_sage_r, nullptr,  xm, 128, 1);

    // ResNet branch: gcn(x,W_res1)->relu->gcn(.,W_res2)
    gemm_wmma<<<G128, 128, 0, stream>>>(x, p_res1, nullptr, xw, 128, 0);
    (void)hipMemsetAsync(agg, 0, NF * sizeof(float), stream);
    edge_agg<<<EW_BLK, 256, 0, stream>>>(xw, ei, dinv, agg, 1);
    gcn_finalize<<<NF_BLK, 256, 0, stream>>>(agg, xw, dinv, b_res1, h1, 1);

    gemm_wmma<<<G128, 128, 0, stream>>>(h1, p_res2, nullptr, xw, 128, 0);
    (void)hipMemsetAsync(agg, 0, NF * sizeof(float), stream);
    edge_agg<<<EW_BLK, 256, 0, stream>>>(xw, ei, dinv, agg, 1);
    gcn_finalize<<<NF_BLK, 256, 0, stream>>>(agg, xw, dinv, b_res2, xgnn, 0);

    // Fusion: out = [xgnn | xm] @ W_fuse + b_fuse  (two K=128 GEMMs)
    gemm_wmma<<<G128, 128, 0, stream>>>(xgnn, p_fuse_t, b_fuse,  out, 128, 0);
    gemm_wmma<<<G128, 128, 0, stream>>>(xm,   p_fuse_b, nullptr, out, 128, 1);
}
